// AFM_B_66030827208976
// MI455X (gfx1250) — compile-verified, matrix operand order checked by reading, not run on hardware
//
#include <hip/hip_runtime.h>
#include <hip/hip_bf16.h>

typedef __attribute__((ext_vector_type(16))) _Float16 v16h;
typedef __attribute__((ext_vector_type(8)))  float    v8f;

struct cpx { float x, y; };

#define PI_F 3.14159265358979323846f
#define MAXR 362.03867196751236f   /* sqrt(512^2+512^2)/2 */

// ---------------------------------------------------------------------------
// 512-point Stockham FFT core: 256 threads, LDS ping-pong, natural order out.
// sign = -1 forward, +1 inverse (unnormalized).
// ---------------------------------------------------------------------------
__device__ __forceinline__ void fft512_core(cpx* b0, cpx* b1, int t, float sign,
                                            cpx& out0, cpx& out1) {
  cpx* src = b0;
  cpx* dst = b1;
  int m = 1;
#pragma unroll
  for (int s = 0; s < 9; ++s) {
    int k  = t & (m - 1);
    int jm = t - k;                       // j*m
    cpx c0 = src[t];
    cpx c1 = src[t + 256];
    float ang = sign * (float)jm * (PI_F / 256.0f);
    float sw, cw;
    __sincosf(ang, &sw, &cw);
    float ex = c0.x - c1.x, ey = c0.y - c1.y;
    cpx d0; d0.x = c0.x + c1.x; d0.y = c0.y + c1.y;
    cpx d1; d1.x = cw * ex - sw * ey; d1.y = cw * ey + sw * ex;
    dst[t + jm]     = d0;
    dst[t + jm + m] = d1;
    __syncthreads();
    cpx* tmp = src; src = dst; dst = tmp;
    m <<= 1;
  }
  out0 = src[t];
  out1 = src[t + 256];
}

// Forward row FFT of real input, with (-1)^(x+y) modulation => shifted spectrum.
__global__ __launch_bounds__(256) void k_fft_rows_fwd(const float* __restrict__ x,
                                                      cpx* __restrict__ spec) {
  __shared__ cpx B0[512];
  __shared__ cpx B1[512];
  int line = blockIdx.x;          // img*512 + y, img = b*3+c
  int y = line & 511;
  int t = threadIdx.x;
  const float* row = x + (size_t)line * 512;
  for (int i = t; i < 512; i += 256) {
    float s = ((i + y) & 1) ? -1.f : 1.f;
    cpx v; v.x = row[i] * s; v.y = 0.f;
    B0[i] = v;
  }
  __syncthreads();
  cpx o0, o1;
  fft512_core(B0, B1, t, -1.f, o0, o1);
  cpx* srow = spec + (size_t)line * 512;
  srow[t] = o0;
  srow[t + 256] = o1;
}

// In-place column FFT (sign selects direction).
__global__ __launch_bounds__(256) void k_fft_cols(cpx* __restrict__ spec, float sign) {
  __shared__ cpx B0[512];
  __shared__ cpx B1[512];
  int img = blockIdx.x >> 9;
  int x   = blockIdx.x & 511;
  cpx* base = spec + ((size_t)img << 18) + x;
  int t = threadIdx.x;
  for (int i = t; i < 512; i += 256) B0[i] = base[(size_t)i << 9];
  __syncthreads();
  cpx o0, o1;
  fft512_core(B0, B1, t, sign, o0, o1);
  base[(size_t)t << 9]         = o0;
  base[(size_t)(t + 256) << 9] = o1;
}

// Inverse row FFT -> real output with (-1)^(x+y)/N^2 factor.
__global__ __launch_bounds__(256) void k_ifft_rows_real(const cpx* __restrict__ spec,
                                                        float* __restrict__ out) {
  __shared__ cpx B0[512];
  __shared__ cpx B1[512];
  int line = blockIdx.x;
  int y = line & 511;
  int t = threadIdx.x;
  const cpx* srow = spec + (size_t)line * 512;
  for (int i = t; i < 512; i += 256) B0[i] = srow[i];
  __syncthreads();
  cpx o0, o1;
  fft512_core(B0, B1, t, 1.f, o0, o1);
  const float sc = 1.f / (512.f * 512.f);
  float s0 = (((t + y) & 1) ? -sc : sc);   // parity(t+256) == parity(t)
  float* orow = out + (size_t)line * 512;
  orow[t]       = o0.x * s0;
  orow[t + 256] = o1.x * s0;
}

// ---------------------------------------------------------------------------
// Build 12-channel f16 CNN input. fftshift rolled batch by 4 and channel by 1,
// so magnitude channels at (b,c) read spectrum image ((b+4)%8, (c+2)%3).
// ---------------------------------------------------------------------------
__global__ __launch_bounds__(256) void k_build_input(const float* __restrict__ noisy,
                                                     const float* __restrict__ clean,
                                                     const cpx* __restrict__ sn,
                                                     const cpx* __restrict__ sc,
                                                     _Float16* __restrict__ fin) {
  size_t i = (size_t)blockIdx.x * 256 + threadIdx.x;   // over 8*512*512
  const size_t HW2 = 512 * 512;
  int b  = (int)(i / HW2);
  size_t pix = i - (size_t)b * HW2;
  for (int c = 0; c < 3; ++c) {
    int rimg = ((b + 4) & 7) * 3 + ((c + 2) % 3);
    fin[((size_t)(b * 12 + c)) * HW2 + pix] =
        (_Float16)noisy[((size_t)(b * 3 + c)) * HW2 + pix];
    cpx vn = sn[(size_t)rimg * HW2 + pix];
    fin[((size_t)(b * 12 + 3 + c)) * HW2 + pix] =
        (_Float16)log10f(sqrtf(vn.x * vn.x + vn.y * vn.y) + 1.f);
    fin[((size_t)(b * 12 + 6 + c)) * HW2 + pix] =
        (_Float16)clean[((size_t)(b * 3 + c)) * HW2 + pix];
    cpx vc = sc[(size_t)rimg * HW2 + pix];
    fin[((size_t)(b * 12 + 9 + c)) * HW2 + pix] =
        (_Float16)log10f(sqrtf(vc.x * vc.x + vc.y * vc.y) + 1.f);
  }
}

// ---------------------------------------------------------------------------
// Implicit-GEMM 3x3 conv with WMMA f32_16x16x32_f16.
// 256 threads = 8 waves; each wave: 16 consecutive x-pixels x 16 out channels.
// K = INC*9 padded to 32*NCHUNK, staged per-chunk through LDS.
// ---------------------------------------------------------------------------
template <int INC, int HW, int NCHUNK, bool F16OUT>
__global__ __launch_bounds__(256) void k_conv3x3(const _Float16* __restrict__ in,
                                                 const float* __restrict__ w,
                                                 const float* __restrict__ bias,
                                                 _Float16* __restrict__ out16,
                                                 float* __restrict__ out32,
                                                 int ntiles) {
  __shared__ _Float16 As[8][16][32];   // per-wave im2col tile (M x Kchunk)
  __shared__ _Float16 Bs[16][32];      // Bt[n][k] = weight(n0+n, k)
  const int tid  = threadIdx.x;
  const int wave = tid >> 5, lane = tid & 31;
  const int half = lane >> 4, row = lane & 15;
  const int b  = blockIdx.z / ntiles;
  const int nt = blockIdx.z % ntiles;
  const int y  = blockIdx.y;
  const int x0 = blockIdx.x * 128 + wave * 16;
  const int n0 = nt * 16;
  const int KV = INC * 9;
  const _Float16* inB = in + (size_t)b * INC * HW * HW;

  v8f acc = {0.f, 0.f, 0.f, 0.f, 0.f, 0.f, 0.f, 0.f};

  for (int kc = 0; kc < NCHUNK; ++kc) {
    __syncthreads();   // previous-iteration fragment reads done
    // ---- stage Bs (whole block): 16x32 = 512 elements
    for (int e = tid; e < 512; e += 256) {
      int n  = e >> 5;
      int kk = e & 31;
      int k  = kc * 32 + kk;
      float v = 0.f;
      if (k < KV) {
        int c = k / 9, r9 = k - c * 9;
        v = w[(((size_t)(n0 + n)) * INC + c) * 9 + r9];
        if (k + 32 < KV) __builtin_prefetch(&w[(((size_t)(n0 + n)) * INC) * 9 + k + 32], 0, 1);
      }
      Bs[n][kk] = (_Float16)v;
    }
    // ---- stage As (per wave): lane owns column kk = lane, 16 pixels
    {
      int k = kc * 32 + lane;
      _Float16 zero = (_Float16)0.f;
      if (k < KV) {
        int c = k / 9, r9 = k - c * 9;
        int dy = r9 / 3, dx = r9 - dy * 3;
        int yy = y + dy - 1;
        bool yok = (yy >= 0) && (yy < HW);
        const _Float16* plane = inB + (size_t)c * HW * HW + (size_t)(yok ? yy : 0) * HW;
        for (int p = 0; p < 16; ++p) {
          int xx = x0 + p + dx - 1;
          _Float16 v = zero;
          if (yok && xx >= 0 && xx < HW) v = plane[xx];
          As[wave][p][lane] = v;
        }
      } else {
        for (int p = 0; p < 16; ++p) As[wave][p][lane] = zero;
      }
    }
    __syncthreads();
    // ---- pack fragments per documented 16-bit A/B layout
    v16h a, bf;
#pragma unroll
    for (int r = 0; r < 8; ++r) {
      int k0 = ((r & 3) * 2) + ((r >> 2) * 16) + half * 8;
      a[2 * r]      = As[wave][row][k0];
      a[2 * r + 1]  = As[wave][row][k0 + 1];
      bf[2 * r]     = Bs[row][k0];
      bf[2 * r + 1] = Bs[row][k0 + 1];
    }
    acc = __builtin_amdgcn_wmma_f32_16x16x32_f16(false, a, false, bf,
                                                 (short)0, acc, false, false);
  }

  // ---- epilogue: D layout => lane n = lane&15, reg r -> pixel m = r + 8*half
  const int outC = ntiles * 16;
  const int n = row;
  float bz = bias[n0 + n];
#pragma unroll
  for (int r = 0; r < 8; ++r) {
    int m = r + 8 * half;
    int x = x0 + m;
    float v = acc[r] + bz;
    v = v > 0.f ? v : 0.f;
    size_t off = (((size_t)b * outC + (n0 + n)) * HW + y) * HW + x;
    if (F16OUT) out16[off] = (_Float16)v;
    else        out32[off] = v;
  }
}

// ---------------------------------------------------------------------------
__global__ __launch_bounds__(256) void k_avgpool2(const _Float16* __restrict__ in,
                                                  _Float16* __restrict__ out,
                                                  int C, int Hi) {
  int Ho = Hi >> 1;
  size_t total = (size_t)8 * C * Ho * Ho;
  size_t i = (size_t)blockIdx.x * 256 + threadIdx.x;
  if (i >= total) return;
  int xo = (int)(i % Ho);
  size_t r = i / Ho;
  int yo = (int)(r % Ho);
  size_t pc = r / Ho;  // b*C + c
  const _Float16* p = in + (pc * Hi + (size_t)yo * 2) * Hi + (size_t)xo * 2;
  float v = (float)p[0] + (float)p[1] + (float)p[Hi] + (float)p[Hi + 1];
  out[i] = (_Float16)(v * 0.25f);
}

__global__ __launch_bounds__(256) void k_meanhw(const float* __restrict__ act,
                                                float* __restrict__ yv) {
  __shared__ float sm[256];
  int pc = blockIdx.x;   // b*64 + c
  const float* p = act + (size_t)pc * 16384;
  float s = 0.f;
  for (int i = threadIdx.x; i < 16384; i += 256) s += p[i];
  sm[threadIdx.x] = s;
  __syncthreads();
  for (int st = 128; st > 0; st >>= 1) {
    if (threadIdx.x < st) sm[threadIdx.x] += sm[threadIdx.x + st];
    __syncthreads();
  }
  if (threadIdx.x == 0) yv[pc] = sm[0] * (1.f / 16384.f);
}

__global__ __launch_bounds__(256) void k_fc(const float* __restrict__ x,
                                            const float* __restrict__ w,
                                            const float* __restrict__ bias,
                                            float* __restrict__ out, int IN, int OUT) {
  int i = blockIdx.x * 256 + threadIdx.x;
  if (i >= 8 * OUT) return;
  int o = i % OUT, b = i / OUT;
  const float* xr = x + (size_t)b * IN;
  const float* wr = w + (size_t)o * IN;
  float s = bias[o];
  for (int k = 0; k < IN; ++k) s += xr[k] * wr[k];
  out[i] = s;
}

// softmax(logits*0.1) over last 100, dotted with radius_factor -> value_set
__global__ __launch_bounds__(128) void k_value(const float* __restrict__ logits,
                                               float* __restrict__ vset) {
  __shared__ float sm[128];
  int rowid = blockIdx.x;   // b*100 + i
  const float* lr = logits + (size_t)rowid * 100;
  int t = threadIdx.x;
  float v = (t < 100) ? lr[t] * 0.1f : -3.4e38f;
  sm[t] = v; __syncthreads();
  for (int st = 64; st > 0; st >>= 1) {
    if (t < st) sm[t] = fmaxf(sm[t], sm[t + st]);
    __syncthreads();
  }
  float mx = sm[0]; __syncthreads();
  float e = (t < 100) ? __expf(v - mx) : 0.f;
  sm[t] = e; __syncthreads();
  for (int st = 64; st > 0; st >>= 1) {
    if (t < st) sm[t] += sm[t + st];
    __syncthreads();
  }
  float ssum = sm[0]; __syncthreads();
  float we = (t < 100) ? e * ((float)(t + 1) * 0.01f) : 0.f;
  sm[t] = we; __syncthreads();
  for (int st = 64; st > 0; st >>= 1) {
    if (t < st) sm[t] += sm[t + st];
    __syncthreads();
  }
  if (t == 0) vset[rowid] = sm[0] / ssum;   // FQ_BOUND == 1
}

// Per-pixel: searchsorted mask value, emit fq_mask, blend spectra in place.
// ifftshift roll means image b is blended with mask[(b+4)%8].
__global__ __launch_bounds__(256) void k_mask_combine(const float* __restrict__ vset,
                                                      cpx* __restrict__ sn,
                                                      cpx* __restrict__ sc,
                                                      float* __restrict__ mask_out) {
  size_t i = (size_t)blockIdx.x * 256 + threadIdx.x;  // over 512*512
  if (i >= 512 * 512) return;
  int x = (int)(i & 511), y = (int)(i >> 9);
  float fy = (float)y - 256.f, fx = (float)x - 256.f;
  float dist = sqrtf(fy * fy + fx * fx);
  int lo = 0, hi = 100;                      // searchsorted(..., 'right')
  while (lo < hi) {
    int mid = (lo + hi) >> 1;
    float rv = MAXR * ((float)(mid + 1) * 0.01f);
    if (rv <= dist) lo = mid + 1; else hi = mid;
  }
  int idx = lo;  // 0..100 (100 -> zero pad)
  float mv[8];
  for (int b = 0; b < 8; ++b) {
    float m = (idx < 100) ? vset[b * 100 + idx] : 0.f;
    mv[b] = m;
    mask_out[(size_t)b * 262144 + i] = m;
  }
  for (int b = 0; b < 8; ++b) {
    float m = mv[(b + 4) & 7];
    for (int c = 0; c < 3; ++c) {
      size_t off = (size_t)(b * 3 + c) * 262144 + i;
      cpx a = sn[off], d = sc[off];
      cpx h, e;
      h.x = a.x * m + d.x * (1.f - m);  h.y = a.y * m + d.y * (1.f - m);
      e.x = a.x * (1.f - m) + d.x * m;  e.y = a.y * (1.f - m) + d.y * m;
      sn[off] = h;   // hard spectrum
      sc[off] = e;   // easy spectrum
    }
  }
}

// ---------------------------------------------------------------------------
extern "C" void kernel_launch(void* const* d_in, const int* in_sizes, int n_in,
                              void* d_out, int out_size, void* d_ws, size_t ws_size,
                              hipStream_t stream) {
  (void)in_sizes; (void)n_in; (void)out_size; (void)ws_size;
  const float* clean   = (const float*)d_in[0];
  const float* noisy   = (const float*)d_in[1];
  const float* conv1_w = (const float*)d_in[2];
  const float* conv1_b = (const float*)d_in[3];
  const float* conv2_w = (const float*)d_in[4];
  const float* conv2_b = (const float*)d_in[5];
  const float* conv3_w = (const float*)d_in[6];
  const float* conv3_b = (const float*)d_in[7];
  const float* fc1_w   = (const float*)d_in[8];
  const float* fc1_b   = (const float*)d_in[9];
  const float* fc2_w   = (const float*)d_in[10];
  const float* fc2_b   = (const float*)d_in[11];
  float* out = (float*)d_out;   // hard | easy | fq_mask

  char* ws = (char*)d_ws;
  cpx*      specN = (cpx*)(ws);                            // 48 MiB
  cpx*      specC = (cpx*)(ws + 50331648ull);              // 48 MiB
  _Float16* fin   = (_Float16*)(ws + 100663296ull);        // 48 MiB (also pool buf)
  char*     actb  = ws + 150994944ull;                     // 64 MiB
  _Float16* act16 = (_Float16*)actb;
  float*    act32 = (float*)actb;
  _Float16* poolb = fin;                                   // reuse after conv1
  float* smallb = (float*)(ws + 218103808ull);
  float* yv     = smallb;          // 512
  float* hbuf   = smallb + 512;    // 2048
  float* logits = hbuf + 2048;     // 80000
  float* vset   = logits + 80000;  // 800

  dim3 blk(256);

  // Forward shifted spectra (parity-modulated FFTs), in place per array.
  k_fft_rows_fwd<<<24 * 512, blk, 0, stream>>>(noisy, specN);
  k_fft_rows_fwd<<<24 * 512, blk, 0, stream>>>(clean, specC);
  k_fft_cols<<<24 * 512, blk, 0, stream>>>(specN, -1.f);
  k_fft_cols<<<24 * 512, blk, 0, stream>>>(specC, -1.f);

  // CNN input (f16, 12 channels).
  k_build_input<<<(8 * 512 * 512) / 256, blk, 0, stream>>>(noisy, clean, specN, specC, fin);

  // conv1: 12->16 @512, K=108 (4 chunks)
  k_conv3x3<12, 512, 4, true><<<dim3(4, 512, 8), blk, 0, stream>>>(
      fin, conv1_w, conv1_b, act16, nullptr, 1);
  k_avgpool2<<<(8 * 16 * 256 * 256 + 255) / 256, blk, 0, stream>>>(act16, poolb, 16, 512);

  // conv2: 16->32 @256, K=144 (5 chunks)
  k_conv3x3<16, 256, 5, true><<<dim3(2, 256, 16), blk, 0, stream>>>(
      poolb, conv2_w, conv2_b, act16, nullptr, 2);
  k_avgpool2<<<(8 * 32 * 128 * 128 + 255) / 256, blk, 0, stream>>>(act16, poolb, 32, 256);

  // conv3: 32->64 @128, K=288 (9 chunks), f32 out for the mean
  k_conv3x3<32, 128, 9, false><<<dim3(1, 128, 32), blk, 0, stream>>>(
      poolb, conv3_w, conv3_b, nullptr, act32, 4);

  k_meanhw<<<512, blk, 0, stream>>>(act32, yv);
  k_fc<<<(8 * 256 + 255) / 256, blk, 0, stream>>>(yv, fc1_w, fc1_b, hbuf, 64, 256);
  k_fc<<<(8 * 10000 + 255) / 256, blk, 0, stream>>>(hbuf, fc2_w, fc2_b, logits, 256, 10000);
  k_value<<<800, dim3(128), 0, stream>>>(logits, vset);

  // Mask + spectral blend (writes fq_mask output, transforms spectra in place).
  k_mask_combine<<<(512 * 512) / 256, blk, 0, stream>>>(vset, specN, specC,
                                                        out + 12582912ull);

  // Inverse transforms: cols then rows(real, parity, 1/N^2).
  k_fft_cols<<<24 * 512, blk, 0, stream>>>(specN, 1.f);
  k_fft_cols<<<24 * 512, blk, 0, stream>>>(specC, 1.f);
  k_ifft_rows_real<<<24 * 512, blk, 0, stream>>>(specN, out);              // hard
  k_ifft_rows_real<<<24 * 512, blk, 0, stream>>>(specC, out + 6291456ull); // easy
}
